// DeepSetModel_72877005078793
// MI455X (gfx1250) — matrix operand order repaired and checked by reading, not compiled, not run on hardware
//
#include <hip/hip_runtime.h>
#include <hip/hip_bf16.h>
#include <math.h>

typedef __attribute__((ext_vector_type(16))) _Float16 v16h;
typedef __attribute__((ext_vector_type(8)))  _Float16 v8h;
typedef __attribute__((ext_vector_type(8)))  float    v8f;

// D = A(16x32 f16) * B(32x16 f16) + C(16x16 f32)
__device__ __forceinline__ v8f wmma_f16(v16h a, v16h b, v8f c) {
  return __builtin_amdgcn_wmma_f32_16x16x32_f16(false, a, false, b, (short)0, c, false, false);
}

// A fragment from row-major LDS tile (leading dim ld halves, ld % 8 == 0).
// ISA layout: lane holds row M=lane&15; lanes 0-15 -> K {0..7,16..23}, lanes 16-31 -> K {8..15,24..31}
__device__ __forceinline__ v16h load_a_frag(const _Float16* buf, int ld, int row0, int k0, int lane) {
  const int row = row0 + (lane & 15);
  const int kb  = k0 + ((lane & 16) ? 8 : 0);
  const _Float16* p = buf + row * ld + kb;        // 16B aligned
  const v8h lo = *(const v8h*)(p);                // K kb..kb+7
  const v8h hi = *(const v8h*)(p + 16);           // K kb+16..kb+23
  return __builtin_shufflevector(lo, hi, 0,1,2,3,4,5,6,7,8,9,10,11,12,13,14,15);
}

// B fragment from pre-swizzled LDS frag buffer: frag*512 + lane*16 halves, contiguous.
__device__ __forceinline__ v16h load_b_fragS(const _Float16* base, int frag, int lane) {
  return *(const v16h*)(base + frag * 512 + lane * 16);   // 32B aligned -> 2x ds_load_b128
}

// Stage a KxN f32 weight matrix into fragment-swizzled f16 LDS.
// frag f = ks*NT + n ; element e of lane l is B[ks*32 + ((l&16)?8:0) + (e&7) + ((e&8)?16:0)][n*16 + (l&15)]
template <int K, int N, int NT, int KS>
__device__ __forceinline__ void stage_b_swizzled(_Float16* dst, const float* __restrict__ W,
                                                 int kvalid, int tid, int nthreads) {
  for (int d = tid; d < KS * NT * 512; d += nthreads) {
    const int f = d >> 9;
    const int rem = d & 511;
    const int l = rem >> 4;
    const int e = rem & 15;
    const int ks = f / NT;
    const int n  = f % NT;
    const int k   = ks * 32 + ((l & 16) ? 8 : 0) + (e & 7) + ((e & 8) ? 16 : 0);
    const int col = n * 16 + (l & 15);
    dst[d] = (k < kvalid) ? (_Float16)W[k * N + col] : (_Float16)0.0f;
  }
}

// ---------------------------------------------------------------------------
// Kernel 1: fused row MLP (20->64->128->64, relu) + segment-sum(20) -> pooled
// Block: 128 threads (4 waves), 80 rows (= 4 segments = 5 row-tiles of 16)
// ---------------------------------------------------------------------------
#define LDA  40    // padded leading dims (halves): 16B-aligned rows, bank-friendly
#define LDH1 72
#define LDH2 136

__global__ __launch_bounds__(128) void deepset_rowmlp_pool(
    const float* __restrict__ X,
    const float* __restrict__ W1, const float* __restrict__ b1,
    const float* __restrict__ W2, const float* __restrict__ b2,
    const float* __restrict__ W3, const float* __restrict__ b3,
    float* __restrict__ pooled, int rows)
{
  __shared__ __align__(32) _Float16 sW1f[4  * 512];   // 4 frags  (K 20->32 padded)
  __shared__ __align__(32) _Float16 sW2f[16 * 512];   // ks*8+n
  __shared__ __align__(32) _Float16 sW3f[16 * 512];   // ks*4+n
  __shared__ float sB1[64], sB2[128], sB3[64];
  __shared__ __align__(16) _Float16 sA [80 * LDA];
  __shared__ __align__(16) _Float16 sH1[80 * LDH1];
  // union: raw X tile (6720B) vs { sH2: 16*136*2 = 4352B , sPool: 256*4 = 1024B }
  __shared__ __align__(16) unsigned char uRaw[6720];
  float*    sXraw = (float*)uRaw;               // 1680 f32, phases A/B only
  _Float16* sH2   = (_Float16*)uRaw;            // 16 x LDH2, phase C
  float*    sPool = (float*)(uRaw + 4352);      // 4 x 64, phase C

  const int tid   = threadIdx.x;
  const int lane  = tid & 31;
  const int wave  = tid >> 5;
  const int rbase = blockIdx.x * 80;
  const bool full = (rbase + 80 <= rows);       // block-uniform

  // --- Phase A: coalesced staging ---
  if (full) {                                   // full tile: 6720B, 16B-aligned float4s
    const float4* Xv = (const float4*)X;
    const int base4 = rbase * 21 / 4;           // rbase*21 is multiple of 4
    float4* Rv = (float4*)sXraw;
    for (int c = tid; c < 420; c += 128) Rv[c] = Xv[base4 + c];
  } else {
    for (int i = tid; i < 1680; i += 128) {
      const long long g = (long long)rbase * 21 + i;
      sXraw[i] = (g < (long long)rows * 21) ? X[g] : 0.0f;
    }
  }
  stage_b_swizzled<32, 64, 4, 1>(sW1f, W1, 20, tid, 128);
  stage_b_swizzled<64, 128, 8, 2>(sW2f, W2, 64, tid, 128);
  stage_b_swizzled<128, 64, 4, 4>(sW3f, W3, 128, tid, 128);
  if (tid < 64)  sB1[tid] = b1[tid];
  if (tid < 128) sB2[tid] = b2[tid];
  if (tid < 64)  sB3[tid] = b3[tid];
  __syncthreads();

  // --- Phase B: convert raw tile -> f16 A tile (skip id col, pad K 20->32) ---
  for (int i = tid; i < 80 * 32; i += 128) {
    const int r = i >> 5, k = i & 31;
    sA[r * LDA + k] = (k < 20) ? (_Float16)sXraw[r * 21 + 1 + k] : (_Float16)0.0f;
  }
  __syncthreads();                              // sXraw dead; uRaw reusable

  // --- Phase C: zero pool accumulators + layer 1 ---
  for (int i = tid; i < 4 * 64; i += 128) sPool[i] = 0.0f;

  const int colw = wave * 16 + (lane & 15);     // wave-owned column (N-tile = wave)
  const int rofs = (lane & 16) ? 8 : 0;

  {                                             // layer 1: wave owns N-tile `wave`, 5 row-tiles
    const v16h bW1 = load_b_fragS(sW1f, wave, lane);
    const float bias = sB1[colw];
#pragma unroll
    for (int t = 0; t < 5; ++t) {
      v8f c = {};
      c = wmma_f16(load_a_frag(sA, LDA, t * 16, 0, lane), bW1, c);
#pragma unroll
      for (int i = 0; i < 8; ++i) {
        float v = c[i] + bias; v = v > 0.0f ? v : 0.0f;
        sH1[(t * 16 + rofs + i) * LDH1 + colw] = (_Float16)v;
      }
    }
  }
  __syncthreads();

  // --- preload loop-invariant B fragments for layers 2 & 3 ---
  v16h bW2[2][2];                               // [ks][nn], N-tiles {wave, wave+4}
#pragma unroll
  for (int ks = 0; ks < 2; ++ks)
#pragma unroll
    for (int nn = 0; nn < 2; ++nn)
      bW2[ks][nn] = load_b_fragS(sW2f, ks * 8 + wave + nn * 4, lane);
  v16h bW3[4];                                  // [ks], N-tile wave
#pragma unroll
  for (int ks = 0; ks < 4; ++ks)
    bW3[ks] = load_b_fragS(sW3f, ks * 4 + wave, lane);
  const float bias2a = sB2[colw];
  const float bias2b = sB2[colw + 64];
  const float bias3  = sB3[colw];

  // --- per row-tile: layer 2 -> layer 3 -> pool ---
  for (int t = 0; t < 5; ++t) {
    {                                           // layer 2: N-tiles {wave, wave+4}
      const v16h a0 = load_a_frag(sH1, LDH1, t * 16, 0,  lane);
      const v16h a1 = load_a_frag(sH1, LDH1, t * 16, 32, lane);
#pragma unroll
      for (int nn = 0; nn < 2; ++nn) {
        v8f c = {};
        c = wmma_f16(a0, bW2[0][nn], c);
        c = wmma_f16(a1, bW2[1][nn], c);
        const int col = colw + nn * 64;
        const float bias = nn ? bias2b : bias2a;
#pragma unroll
        for (int i = 0; i < 8; ++i) {
          float v = c[i] + bias; v = v > 0.0f ? v : 0.0f;
          sH2[(rofs + i) * LDH2 + col] = (_Float16)v;
        }
      }
    }
    __syncthreads();

    {                                           // layer 3: N-tile `wave`
      v8f c = {};
#pragma unroll
      for (int ks = 0; ks < 4; ++ks)
        c = wmma_f16(load_a_frag(sH2, LDH2, 0, ks * 32, lane), bW3[ks], c);
      if (full) {
#pragma unroll
        for (int i = 0; i < 8; ++i) {
          const int lr = t * 16 + rofs + i;     // local row in [0,80)
          float v = c[i] + bias3; v = v > 0.0f ? v : 0.0f;
          atomicAdd(&sPool[(lr / 20) * 64 + colw], v);   // ds_add_f32
        }
      } else {
#pragma unroll
        for (int i = 0; i < 8; ++i) {
          const int lr = t * 16 + rofs + i;
          if (rbase + lr < rows) {
            float v = c[i] + bias3; v = v > 0.0f ? v : 0.0f;
            atomicAdd(&sPool[(lr / 20) * 64 + colw], v);
          }
        }
      }
    }
    __syncthreads();
  }

  // --- emit pooled segments (contiguous 1KB per block, float4 stores) ---
  if (tid < 64) {
    const int s = tid >> 4, q = tid & 15;
    const int gs = blockIdx.x * 4 + s;
    if (gs * 20 < rows)
      ((float4*)pooled)[(long long)gs * 16 + q] = ((const float4*)sPool)[s * 16 + q];
  }
}

// ---------------------------------------------------------------------------
// Kernel 2: segment MLP (64->128 relu ->64 relu ->1 sigmoid) on pooled rows
// Block: 128 threads (4 waves), 64 segments (4 row-tiles of 16)
// ---------------------------------------------------------------------------
#define LDP  72
#define LDG1 136

__global__ __launch_bounds__(128) void deepset_segmlp(
    const float* __restrict__ pooled,
    const float* __restrict__ R1, const float* __restrict__ rb1,
    const float* __restrict__ R2, const float* __restrict__ rb2,
    const float* __restrict__ R3, const float* __restrict__ rb3,
    float* __restrict__ out, int nseg)
{
  __shared__ __align__(32) _Float16 sR1f[16 * 512];   // ks*8+n
  __shared__ __align__(32) _Float16 sR2f[16 * 512];   // ks*4+n
  __shared__ __align__(16) _Float16 sP[64 * LDP];
  // union: sG1 (64*136*2 = 17408B f16) vs r2buf (64*64*4 = 16384B f32)
  __shared__ __align__(16) unsigned char uBuf[64 * LDG1 * 2];
  _Float16* sG1   = (_Float16*)uBuf;
  float*    r2buf = (float*)uBuf;
  __shared__ float sRb1[128], sRb2[64], sR3v[64];
  __shared__ float sRb3;

  const int tid   = threadIdx.x;
  const int lane  = tid & 31;
  const int wave  = tid >> 5;
  const int rbase = blockIdx.x * 64;

  stage_b_swizzled<64, 128, 8, 2>(sR1f, R1, 64, tid, 128);
  stage_b_swizzled<128, 64, 4, 4>(sR2f, R2, 128, tid, 128);
  if (tid < 128) sRb1[tid] = rb1[tid];
  if (tid < 64)  sRb2[tid] = rb2[tid];
  if (tid < 64)  sR3v[tid] = R3[tid];
  if (tid == 0)  sRb3 = rb3[0];
  // pooled tile: 64 rows x 64 f32, row-contiguous -> float4 loads
  for (int i4 = tid; i4 < 64 * 16; i4 += 128) {
    const int r = i4 >> 4, c4 = (i4 & 15) * 4;
    const int g = rbase + r;
    float4 v = make_float4(0.f, 0.f, 0.f, 0.f);
    if (g < nseg) v = ((const float4*)pooled)[(long long)g * 16 + (i4 & 15)];
    _Float16* p = &sP[r * LDP + c4];
    p[0] = (_Float16)v.x; p[1] = (_Float16)v.y; p[2] = (_Float16)v.z; p[3] = (_Float16)v.w;
  }
  __syncthreads();

  const int colw = wave * 16 + (lane & 15);
  const int rofs = (lane & 16) ? 8 : 0;

  // R1: wave owns N-tiles {wave, wave+4} for all 4 row-tiles; K=64 -> 2 k-steps
  {
    v16h bR1[2][2];                             // [ks][nn]
#pragma unroll
    for (int ks = 0; ks < 2; ++ks)
#pragma unroll
      for (int nn = 0; nn < 2; ++nn)
        bR1[ks][nn] = load_b_fragS(sR1f, ks * 8 + wave + nn * 4, lane);
    const float biasA = sRb1[colw];
    const float biasB = sRb1[colw + 64];
#pragma unroll
    for (int t = 0; t < 4; ++t) {
      const v16h a0 = load_a_frag(sP, LDP, t * 16, 0,  lane);
      const v16h a1 = load_a_frag(sP, LDP, t * 16, 32, lane);
#pragma unroll
      for (int nn = 0; nn < 2; ++nn) {
        v8f c = {};
        c = wmma_f16(a0, bR1[0][nn], c);
        c = wmma_f16(a1, bR1[1][nn], c);
        const int col = colw + nn * 64;
        const float bias = nn ? biasB : biasA;
#pragma unroll
        for (int i = 0; i < 8; ++i) {
          float v = c[i] + bias; v = v > 0.0f ? v : 0.0f;
          sG1[(t * 16 + rofs + i) * LDG1 + col] = (_Float16)v;
        }
      }
    }
  }
  __syncthreads();

  // R2: wave handles row-tile t=wave, all 4 N-tiles; K=128 -> 4 k-steps
  v16h aT[4];
#pragma unroll
  for (int ks = 0; ks < 4; ++ks) aT[ks] = load_a_frag(sG1, LDG1, wave * 16, ks * 32, lane);
  v8f acc[4];
#pragma unroll
  for (int n = 0; n < 4; ++n) {
    v8f c = {};
#pragma unroll
    for (int ks = 0; ks < 4; ++ks)
      c = wmma_f16(aT[ks], load_b_fragS(sR2f, ks * 4 + n, lane), c);
    acc[n] = c;
  }
  __syncthreads();   // all reads of sG1 done; safe to overwrite as r2buf

#pragma unroll
  for (int n = 0; n < 4; ++n) {
    const int col = n * 16 + (lane & 15);
    const float bias = sRb2[col];
#pragma unroll
    for (int i = 0; i < 8; ++i) {
      float v = acc[n][i] + bias; v = v > 0.0f ? v : 0.0f;
      r2buf[(wave * 16 + rofs + i) * 64 + col] = v;
    }
  }
  __syncthreads();

  // final: y = sigmoid(r2 . R3 + rb3), one segment per thread (tid < 64)
  if (tid < 64) {
    const int g = rbase + tid;
    if (g < nseg) {
      float s = sRb3;
#pragma unroll 8
      for (int c = 0; c < 64; ++c) s += r2buf[tid * 64 + c] * sR3v[c];
      out[g] = 1.0f / (1.0f + __expf(-s));
    }
  }
}

// ---------------------------------------------------------------------------
extern "C" void kernel_launch(void* const* d_in, const int* in_sizes, int n_in,
                              void* d_out, int out_size, void* d_ws, size_t ws_size,
                              hipStream_t stream) {
  const float* X   = (const float*)d_in[0];
  const float* W1  = (const float*)d_in[1];
  const float* b1  = (const float*)d_in[2];
  const float* W2  = (const float*)d_in[3];
  const float* b2  = (const float*)d_in[4];
  const float* W3  = (const float*)d_in[5];
  const float* b3  = (const float*)d_in[6];
  const float* R1  = (const float*)d_in[7];
  const float* rb1 = (const float*)d_in[8];
  const float* R2  = (const float*)d_in[9];
  const float* rb2 = (const float*)d_in[10];
  const float* R3  = (const float*)d_in[11];
  const float* rb3 = (const float*)d_in[12];

  const int rows = in_sizes[0] / 21;   // 1,000,000
  const int nseg = rows / 20;          // 50,000 (matches reference n_segments)

  float* pooled = (float*)d_ws;        // nseg * 64 f32 = 12.8 MB scratch

  const int blocks1 = (rows + 79) / 80;
  deepset_rowmlp_pool<<<blocks1, 128, 0, stream>>>(X, W1, b1, W2, b2, W3, b3,
                                                   pooled, rows);

  const int blocks2 = (nseg + 63) / 64;
  deepset_segmlp<<<blocks2, 128, 0, stream>>>(pooled, R1, rb1, R2, rb2, R3, rb3,
                                              (float*)d_out, nseg);
}
